// FuseFormer_6588479832658
// MI455X (gfx1250) — matrix operand order repaired and checked by reading, not compiled, not run on hardware
//
#include <hip/hip_runtime.h>
#include <hip/hip_bf16.h>
#include <math.h>

typedef _Float16 h16;
typedef __attribute__((ext_vector_type(16))) _Float16 v16h;
typedef __attribute__((ext_vector_type(8)))  _Float16 v8h;
typedef __attribute__((ext_vector_type(8)))  float    v8f;

// load 8 halves (16B) into elements [base, base+8) of a v16h
__device__ __forceinline__ void ld8(v16h& d, int base, const h16* p) {
    v8h v = *(const v8h*)p;
    #pragma unroll
    for (int e = 0; e < 8; ++e) d[base + e] = v[e];
}
__device__ __forceinline__ v16h ld16(const h16* p) {
    v16h r;
    ld8(r, 0, p);
    ld8(r, 8, p + 8);
    return r;
}

// ---------------------------------------------------------------- zero fill (pads must be 0 for WMMA K=64)
__global__ void k_zero(uint4* __restrict__ p, int n16) {
    int t = blockIdx.x * blockDim.x + threadIdx.x;
    if (t < n16) p[t] = uint4{0u, 0u, 0u, 0u};
}

// ---------------------------------------------------------------- bicubic x4
__device__ __forceinline__ float cubicw(float t) {
    const float A = -0.75f;
    t = fabsf(t);
    float t2 = t * t, t3 = t2 * t;
    if (t <= 1.f) return (A + 2.f) * t3 - (A + 3.f) * t2 + 1.f;
    if (t < 2.f)  return A * t3 - 5.f * A * t2 + 8.f * A * t - 4.f * A;
    return 0.f;
}

// up[b][c][64][64] <- HSI[b][c][16][16], torch bicubic (align_corners=False, edge clamp), clip [0,1]
__global__ void k_upsample(const float* __restrict__ hsi, float* __restrict__ up) {
    int t = blockIdx.x * blockDim.x + threadIdx.x;      // 4*31*64*64 = 507904
    int j = t & 63, i = (t >> 6) & 63, bc = t >> 12;    // bc in [0,124)
    float yy = (i + 0.5f) * 0.25f - 0.5f;
    float xx = (j + 0.5f) * 0.25f - 0.5f;
    int iy0 = (int)floorf(yy), ix0 = (int)floorf(xx);
    const float* src = hsi + bc * 256;
    float acc = 0.f;
    for (int dy = -1; dy <= 2; ++dy) {
        int sy = iy0 + dy;
        float wy = cubicw(yy - (float)sy);
        sy = min(max(sy, 0), 15);
        for (int dx = -1; dx <= 2; ++dx) {
            int sx = ix0 + dx;
            float wx = cubicw(xx - (float)sx);
            sx = min(max(sx, 0), 15);
            acc += wy * wx * src[sy * 16 + sx];
        }
    }
    up[t] = fminf(fmaxf(acc, 0.f), 1.f);
}

// ---------------------------------------------------------------- embed 34->48
__global__ void k_embed(const float* __restrict__ up, const float* __restrict__ msi,
                        const float* __restrict__ W, const float* __restrict__ bias,
                        float* __restrict__ x) {
    int t = blockIdx.x * blockDim.x + threadIdx.x;      // 16384*48
    int o = t % 48, row = t / 48;
    int b = row >> 12, n = row & 4095;
    float acc = bias[o];
    const float* upb = up  + b * 31 * 4096 + n;
    const float* msb = msi + b * 3 * 4096 + n;
    #pragma unroll 1
    for (int c = 0; c < 31; ++c) acc += upb[c * 4096] * W[c * 48 + o];
    for (int c = 0; c < 3;  ++c) acc += msb[c * 4096] * W[(31 + c) * 48 + o];
    x[t] = acc;
}

// ---------------------------------------------------------------- layernorm f32[.,48] -> f16[.,64] (cols 48..63 stay 0)
__global__ void k_ln(const float* __restrict__ x, const float* __restrict__ g,
                     const float* __restrict__ be, h16* __restrict__ xn, int M) {
    int r = blockIdx.x * blockDim.x + threadIdx.x;
    if (r >= M) return;
    const float* xr = x + r * 48;
    float s = 0.f, s2 = 0.f;
    for (int i = 0; i < 48; ++i) { float v = xr[i]; s += v; s2 += v * v; }
    float mu  = s * (1.f / 48.f);
    float var = s2 * (1.f / 48.f) - mu * mu;
    float inv = rsqrtf(var + 1e-5f);
    h16* o = xn + (size_t)r * 64;
    for (int i = 0; i < 48; ++i) o[i] = (h16)((xr[i] - mu) * inv * g[i] + be[i]);
}

// ---------------------------------------------------------------- weight f32[K=48][N] -> f16 transposed [N][64] (pad pre-zeroed)
__global__ void k_wcvt(const float* __restrict__ src, h16* __restrict__ dst, int N) {
    int t = blockIdx.x * blockDim.x + threadIdx.x;
    if (t >= 48 * N) return;
    int n = t % N, k = t / N;
    dst[n * 64 + k] = (h16)src[k * N + n];
}

// ---------------------------------------------------------------- WMMA GEMM: C[M,N] = A[M,48] * B[48,N], K padded to 64
// A: f16, row stride lda (=64, cols 48..63 zero).  BT: [N][64] f16, rows 48..63 zero.
// One wave per 16x16 tile, two 16x16x32 WMMAs (K=64), all loads unconditional b128.
template<bool BIAS, bool RESID, bool LEAKY, bool OF, bool OH>
__global__ __launch_bounds__(128)
void k_gemm(const h16* __restrict__ A, int lda, const h16* __restrict__ BT,
            const float* __restrict__ bias, const float* resid, int ldr,
            float* outF, h16* outH, int ldo, int N) {
    int wave = blockIdx.x * (blockDim.x >> 5) + (threadIdx.x >> 5);
    int lane = threadIdx.x & 31;
    int hs = lane >> 4, l = lane & 15;
    int tN = N >> 4;
    int m0 = (wave / tN) << 4;
    int n0 = (wave % tN) << 4;
    v8f c = {};
    const h16* arow = A  + (size_t)(m0 + l) * lda;  // A row m0+l (both halves share the row)
    const h16* bcol = BT + (size_t)(n0 + l) * 64;   // B column n0+l
    #pragma unroll
    for (int k0 = 0; k0 < 64; k0 += 32) {
        v16h a, b;
        a = ld16(arow + k0 + 16 * hs);              // A: K = k0+16hs+e
        b = ld16(bcol + k0 + 16 * hs);              // B: lane holds column l, K = k0+16hs+e
        c = __builtin_amdgcn_wmma_f32_16x16x32_f16(false, a, false, b, (short)0, c, false, false);
    }
    #pragma unroll
    for (int r = 0; r < 8; ++r) {                   // C: lane(hs,l) holds rows 8hs+r, col l
        int row = m0 + 8 * hs + r;
        int col = n0 + l;
        float v = c[r];
        if (BIAS)  v += bias[col];
        if (RESID) v += resid[(size_t)row * ldr + col];
        if (LEAKY) v = (v > 0.f) ? v : 0.01f * v;
        if (OF) outF[(size_t)row * ldo + col] = v;
        if (OH) outH[(size_t)row * ldo + col] = (h16)v;
    }
}

// ---------------------------------------------------------------- qkv[row][144]f16 -> Q[bh][n][16], K[bh][n][16], Vt[bh][16][n]
__global__ void k_pack(const h16* __restrict__ qkv, h16* __restrict__ Qh,
                       h16* __restrict__ Kh, h16* __restrict__ Vt) {
    int t = blockIdx.x * blockDim.x + threadIdx.x;  // 12*4096
    int n = t & 4095, bh = t >> 12;
    int b = bh / 3, h = bh % 3;
    const h16* src = qkv + (size_t)(b * 4096 + n) * 144 + h * 16;
    h16* q = Qh + (size_t)(bh * 4096 + n) * 16;
    h16* k = Kh + (size_t)(bh * 4096 + n) * 16;
    #pragma unroll
    for (int d = 0; d < 16; ++d) {
        q[d] = src[d];
        k[d] = src[48 + d];
        Vt[(size_t)(bh * 16 + d) * 4096 + n] = src[96 + d];
    }
}

// ---------------------------------------------------------------- flash attention, one wave per 16-query tile
// S tile goes through LDS once (C-layout -> row-per-lane); softmax is in-lane VALU +
// one shfl_xor(16) per reduction; P fragment is built in-register in A-layout order.
__global__ __launch_bounds__(128)
void k_flash(const h16* __restrict__ Qh, const h16* __restrict__ Kh,
             const h16* __restrict__ Vt, h16* __restrict__ O) {
    __shared__ float slds[4][16][36];               // 16x32 S tile, pitch 36 (16B-aligned, conflict-free)
    int wv   = threadIdx.x >> 5;
    int wave = blockIdx.x * 4 + wv;
    int lane = threadIdx.x & 31;
    int hs = lane >> 4, l = lane & 15;
    int bh = wave >> 8;                             // 256 query tiles per (b,h)
    int q0 = (wave & 255) << 4;
    int b = bh / 3, h = bh % 3;

    // Q fragment (A layout, d=16 zero-padded to K=32)
    v16h qa = {};
    if (hs == 0) qa = ld16(Qh + (size_t)(bh * 4096 + q0 + l) * 16);

    float mrow = -1e30f, lrow = 0.f;                // softmax state for query row l (dup in both halves)
    v8f acc = {};
    const h16* Kb = Kh + (size_t)bh * 4096 * 16;
    const h16* Vb = Vt + (size_t)bh * 16 * 4096 + (size_t)l * 4096;
    const v8f zc = {};

    for (int jb = 0; jb < 4096; jb += 32) {
        // B fragments for S = Q*K^T: lane(hs=0,l) holds column l (key jb+l / jb+16+l), dims 0..15
        v16h b0 = {}, b1 = {};
        if (hs == 0) {
            b0 = ld16(Kb + (size_t)(jb + l) * 16);
            b1 = ld16(Kb + (size_t)(jb + 16 + l) * 16);
        }
        v8f s0 = __builtin_amdgcn_wmma_f32_16x16x32_f16(false, qa, false, b0, (short)0, zc, false, false);
        v8f s1 = __builtin_amdgcn_wmma_f32_16x16x32_f16(false, qa, false, b1, (short)0, zc, false, false);

        // transpose S to row-per-lane via LDS (apply SCALE = 16^-0.5 here)
        #pragma unroll
        for (int r = 0; r < 8; ++r) {               // C layout: lane(hs,l) holds rows 8hs+r, col l
            slds[wv][8 * hs + r][l]      = s0[r] * 0.25f;
            slds[wv][8 * hs + r][16 + l] = s1[r] * 0.25f;
        }
        __asm__ volatile("s_wait_dscnt 0x0" ::: "memory");  // S stores visible before row reads

        // read back row l, elements already in A-fragment order
        float sv[16];
        #pragma unroll
        for (int e = 0; e < 8; ++e) {
            sv[e]     = slds[wv][l][16 * hs + e];
            sv[8 + e] = slds[wv][l][16 * hs + 8 + e];
        }

        // online softmax, in-lane over 16 elems + one cross-half combine
        float mx = sv[0];
        #pragma unroll
        for (int e = 1; e < 16; ++e) mx = fmaxf(mx, sv[e]);
        mx = fmaxf(mx, __shfl_xor(mx, 16));
        float mn   = fmaxf(mrow, mx);
        float corr = __expf(mrow - mn);
        mrow = mn;
        v16h pa;
        float rs = 0.f;
        #pragma unroll
        for (int e = 0; e < 16; ++e) {
            float p = __expf(sv[e] - mn);
            rs += p;
            pa[e] = (h16)p;
        }
        rs += __shfl_xor(rs, 16);
        lrow = lrow * corr + rs;

        // rescale O accumulator: C-layout row 8hs+r's corr lives in lane 8hs+r
        #pragma unroll
        for (int r = 0; r < 8; ++r) acc[r] *= __shfl(corr, 8 * hs + r);

        // V fragment (B layout): lane(hs,l) = dim-column l, keys jb+16hs+0..15 (contiguous in Vt)
        v16h vb = ld16(Vb + jb + 16 * hs);
        acc = __builtin_amdgcn_wmma_f32_16x16x32_f16(false, pa, false, vb, (short)0, acc, false, false);
        __asm__ volatile("s_wait_dscnt 0x0" ::: "memory");  // WAR vs next iteration's S stores
    }
    #pragma unroll
    for (int r = 0; r < 8; ++r) {
        float li = __shfl(lrow, 8 * hs + r);
        int row = b * 4096 + q0 + 8 * hs + r;
        O[(size_t)row * 64 + h * 16 + l] = (h16)(acc[r] / li);   // attn buffer stride 64
    }
}

// ---------------------------------------------------------------- 3x3 conv 48->48 + leaky (channels-last in/out)
__global__ void k_conv48(const float* __restrict__ in, const float* __restrict__ W,
                         const float* __restrict__ bias, float* __restrict__ out) {
    int t = blockIdx.x * blockDim.x + threadIdx.x;  // 4*48*4096
    int x = t & 63, y = (t >> 6) & 63;
    int rem = t >> 12;
    int co = rem % 48, b = rem / 48;
    float acc = bias[co];
    for (int dy = 0; dy < 3; ++dy) {
        int iy = y + dy - 1;
        if ((unsigned)iy >= 64u) continue;
        for (int dx = 0; dx < 3; ++dx) {
            int ix = x + dx - 1;
            if ((unsigned)ix >= 64u) continue;
            const float* ip = in + (size_t)((b << 12) + iy * 64 + ix) * 48;
            const float* wp = W + co * 432 + dy * 3 + dx;
            for (int ci = 0; ci < 48; ++ci) acc += ip[ci] * wp[ci * 9];
        }
    }
    out[(size_t)((b << 12) + y * 64 + x) * 48 + co] = (acc > 0.f) ? acc : 0.01f * acc;
}

// ---------------------------------------------------------------- 3x3 conv 48->31 + up residual + clip -> d_out
__global__ void k_conv_out(const float* __restrict__ in, const float* __restrict__ W,
                           const float* __restrict__ bias, const float* __restrict__ up,
                           float* __restrict__ out) {
    int t = blockIdx.x * blockDim.x + threadIdx.x;  // 4*31*4096
    int x = t & 63, y = (t >> 6) & 63;
    int rem = t >> 12;
    int co = rem % 31, b = rem / 31;
    float acc = bias[co];
    for (int dy = 0; dy < 3; ++dy) {
        int iy = y + dy - 1;
        if ((unsigned)iy >= 64u) continue;
        for (int dx = 0; dx < 3; ++dx) {
            int ix = x + dx - 1;
            if ((unsigned)ix >= 64u) continue;
            const float* ip = in + (size_t)((b << 12) + iy * 64 + ix) * 48;
            const float* wp = W + co * 432 + dy * 3 + dx;
            for (int ci = 0; ci < 48; ++ci) acc += ip[ci] * wp[ci * 9];
        }
    }
    int oidx = ((b * 31 + co) * 64 + y) * 64 + x;
    float v = acc + up[oidx];
    out[oidx] = fminf(fmaxf(v, 0.f), 1.f);
}

// ================================================================ host
extern "C" void kernel_launch(void* const* d_in, const int* in_sizes, int n_in,
                              void* d_out, int out_size, void* d_ws, size_t ws_size,
                              hipStream_t stream) {
    (void)in_sizes; (void)n_in; (void)out_size; (void)ws_size;
    const int M = 16384;
    char* ws = (char*)d_ws;
    size_t off = 0;
    auto alloc = [&](size_t bytes) -> char* {
        char* p = ws + off;
        off = (off + bytes + 255) & ~(size_t)255;
        return p;
    };
    float* up    = (float*)alloc((size_t)4 * 31 * 4096 * 4);
    float* x     = (float*)alloc((size_t)M * 48 * 4);
    h16*   qkvh  = (h16*)  alloc((size_t)M * 144 * 2);
    h16*   Qh    = (h16*)  alloc((size_t)12 * 4096 * 16 * 2);
    h16*   Kh    = (h16*)  alloc((size_t)12 * 4096 * 16 * 2);
    h16*   Vt    = (h16*)  alloc((size_t)12 * 16 * 4096 * 2);
    float* h1    = (float*)alloc((size_t)M * 48 * 4);
    // --- zero-padded region (K padded 48->64): zeroed once per launch, pads never rewritten
    char*  zbase = ws + off;
    h16*   xn    = (h16*)  alloc((size_t)M * 64 * 2);      // LN out, stride 64
    h16*   attnh = (h16*)  alloc((size_t)M * 64 * 2);      // attention out, stride 64
    h16*   hf    = (h16*)  alloc((size_t)M * 64 * 2);      // FF hidden, stride 64
    h16*   whbase= (h16*)  alloc((size_t)1536 * 64 * 2);   // transposed f16 weights [N][64]
    size_t zbytes = (size_t)((ws + off) - zbase);

    // ---- parse inputs (setup_inputs() insertion order, nested dicts in insertion order)
    int ip = 0;
    auto getp = [&]() { return (const float*)d_in[ip++]; };
    const float* HSI  = getp();
    const float* MSI  = getp();
    const float* embW = getp();
    const float* embB = getp();

    struct Op {
        int kind;                      // 0 = attn (W1=Wqkv no bias, W2=Wo, b2=bo); 1 = ff
        const float *g, *be, *W1, *b1, *W2, *b2;
        h16 *W1h, *W2h;
    };
    Op ops[10]; int nops = 0;
    for (int ly = 0; ly < 2; ++ly) {   // Transformer_E layers: ln1, attn1, ln2, ff
        const float *g1 = getp(), *bb1 = getp();
        const float *Wqkv = getp(), *Wo = getp(), *bo = getp();
        const float *g2 = getp(), *bb2 = getp();
        const float *W1 = getp(), *fb1 = getp(), *W2 = getp(), *fb2 = getp();
        ops[nops++] = {0, g1, bb1, Wqkv, nullptr, Wo, bo, nullptr, nullptr};
        ops[nops++] = {1, g2, bb2, W1, fb1, W2, fb2, nullptr, nullptr};
    }
    for (int ly = 0; ly < 2; ++ly) {   // Transformer_D layers: ln1, attn1, ln2, attn2, ln3, ff
        const float *g1 = getp(), *bb1 = getp();
        const float *Wq1 = getp(), *Wo1 = getp(), *bo1 = getp();
        const float *g2 = getp(), *bb2 = getp();
        const float *Wq2 = getp(), *Wo2 = getp(), *bo2 = getp();
        const float *g3 = getp(), *bb3 = getp();
        const float *W1 = getp(), *fb1 = getp(), *W2 = getp(), *fb2 = getp();
        ops[nops++] = {0, g1, bb1, Wq1, nullptr, Wo1, bo1, nullptr, nullptr};
        ops[nops++] = {0, g2, bb2, Wq2, nullptr, Wo2, bo2, nullptr, nullptr};
        ops[nops++] = {1, g3, bb3, W1, fb1, W2, fb2, nullptr, nullptr};
    }
    const float* c1W = getp(); const float* c1b = getp();
    const float* c2W = getp(); const float* c2b = getp();

    // ---- zero the padded region, then convert weights to f16 [N][64]
    {
        int n16 = (int)(zbytes / 16);
        k_zero<<<(n16 + 255) / 256, 256, 0, stream>>>((uint4*)zbase, n16);
    }
    h16* wp = whbase;
    for (int i = 0; i < nops; ++i) {
        Op& o = ops[i];
        int N1 = (o.kind == 0) ? 144 : 48;
        o.W1h = wp; wp += (size_t)N1 * 64;
        k_wcvt<<<(48 * N1 + 255) / 256, 256, 0, stream>>>(o.W1, o.W1h, N1);
        o.W2h = wp; wp += (size_t)48 * 64;
        k_wcvt<<<9, 256, 0, stream>>>(o.W2, o.W2h, 48);
    }

    // ---- pipeline
    k_upsample<<<1984, 256, 0, stream>>>(HSI, up);
    k_embed<<<3072, 256, 0, stream>>>(up, MSI, embW, embB, x);

    for (int i = 0; i < nops; ++i) {
        Op& o = ops[i];
        k_ln<<<64, 256, 0, stream>>>(x, o.g, o.be, xn, M);
        if (o.kind == 0) {
            // qkv: [16384,48] x [48,144] -> f16 [row][144]
            k_gemm<false, false, false, false, true>
                <<<(M / 16) * 9 / 4, 128, 0, stream>>>(xn, 64, o.W1h, nullptr, nullptr, 0,
                                                       nullptr, qkvh, 144, 144);
            k_pack<<<192, 256, 0, stream>>>(qkvh, Qh, Kh, Vt);
            k_flash<<<768, 128, 0, stream>>>(Qh, Kh, Vt, attnh);
            // out proj + bo + residual -> x
            k_gemm<true, true, false, true, false>
                <<<(M / 16) * 3 / 4, 128, 0, stream>>>(attnh, 64, o.W2h, o.b2, x, 48,
                                                       x, nullptr, 48, 48);
        } else {
            k_gemm<true, false, true, false, true>
                <<<(M / 16) * 3 / 4, 128, 0, stream>>>(xn, 64, o.W1h, o.b1, nullptr, 0,
                                                       nullptr, hf, 64, 48);
            k_gemm<true, true, false, true, false>
                <<<(M / 16) * 3 / 4, 128, 0, stream>>>(hf, 64, o.W2h, o.b2, x, 48,
                                                       x, nullptr, 48, 48);
        }
    }

    k_conv48<<<3072, 256, 0, stream>>>(x, c1W, c1b, h1);
    k_conv_out<<<1984, 256, 0, stream>>>(h1, c2W, c2b, up, (float*)d_out);
}
